// SentenceEncoder_6863357739450
// MI455X (gfx1250) — matrix operand order repaired and checked by reading, not compiled, not run on hardware
//
#include <hip/hip_runtime.h>
#include <hip/hip_bf16.h>
#include <stdint.h>

typedef __bf16 v16bf __attribute__((ext_vector_type(16)));
typedef float  v8f   __attribute__((ext_vector_type(8)));

#define S_LEN   64
#define E_DIM   256
#define H_DIM   512
#define G_DIM   2048
#define ROWS    16
#define D_DOCS  32
#define OUT_STRIDE 1024

#define IH_GSTRIDE (H_DIM * E_DIM * 2)   // 262144 B between gates in w_ih
#define HH_GSTRIDE (H_DIM * H_DIM * 2)   // 524288 B between gates in w_hh

__device__ __forceinline__ unsigned short f2bf(float f) {
  union { float f; uint32_t u; } v; v.f = f;
  uint32_t u = v.u;
  uint32_t r = u + 0x7FFFu + ((u >> 16) & 1u);   // round-to-nearest-even
  if ((u & 0x7F800000u) == 0x7F800000u) r = u;   // inf/nan passthrough
  return (unsigned short)(r >> 16);
}

__device__ __forceinline__ float fsigmoid(float x) {
  return 1.0f / (1.0f + __expf(-x));
}
__device__ __forceinline__ float ftanh_fast(float x) {
  float e = __expf(2.0f * x);
  return (e - 1.0f) / (e + 1.0f);
}

union AF { uint4 q[2]; v16bf v; };
union CF { v8f v; float f[8]; };

// ---------------------------------------------------------------- weights -> bf16
__global__ void cvt_bf16_kernel(const float* __restrict__ src,
                                unsigned short* __restrict__ dst, int n) {
  int i = blockIdx.x * blockDim.x + threadIdx.x;
  if (i < n) dst[i] = f2bf(src[i]);
}

// ---------------------------------------------------------------- embedding gather -> bf16
// one block = 8 tokens, 32 threads per token, 8 floats per thread
__global__ void embed_kernel(const int* __restrict__ docs,
                             const float* __restrict__ embed,
                             unsigned short* __restrict__ xbf) {
  int tok = blockIdx.x * 8 + (threadIdx.x >> 5);
  int e0  = (threadIdx.x & 31) * 8;
  int t = docs[tok];
  const float4* src = reinterpret_cast<const float4*>(embed + (size_t)t * E_DIM + e0);
  float4 a = src[0], b = src[1];
  unsigned short h[8] = { f2bf(a.x), f2bf(a.y), f2bf(a.z), f2bf(a.w),
                          f2bf(b.x), f2bf(b.y), f2bf(b.z), f2bf(b.w) };
  *reinterpret_cast<uint4*>(xbf + (size_t)tok * E_DIM + e0) =
      *reinterpret_cast<uint4*>(h);
}

// ---------------------------------------------------------------- fused BiLSTM + pool
// grid.x = 64 row tiles (16 sequences each), grid.y = direction (0 fwd, 1 bwd)
// block = 256 threads = 8 waves; wave w owns hidden tiles [4w, 4w+4)
__launch_bounds__(256)
__global__ void lstm_kernel(const unsigned short* __restrict__ xbf,
                            const int* __restrict__ docs,
                            const int* __restrict__ doc_lens,
                            const unsigned short* __restrict__ wih_f,
                            const unsigned short* __restrict__ whh_f,
                            const unsigned short* __restrict__ wih_b,
                            const unsigned short* __restrict__ whh_b,
                            const float* __restrict__ b_f,
                            const float* __restrict__ b_b,
                            float* __restrict__ out) {
  const int dir  = blockIdx.y;
  const int base = blockIdx.x * ROWS;
  const unsigned short* wih  = dir ? wih_b : wih_f;
  const unsigned short* whh  = dir ? whh_b : whh_f;
  const float*          bias = dir ? b_b   : b_f;

  __shared__ __align__(16) unsigned short xs[ROWS][E_DIM];       // 8 KB: x_t tile
  __shared__ __align__(16) unsigned short hsb[2][ROWS][H_DIM];   // 32 KB: h, double buffered
  __shared__ __align__(16) float ps[ROWS][H_DIM];                // 32 KB: pooled-h accum
  __shared__ int slen[ROWS];

  const int tid   = threadIdx.x;
  const int wave  = tid >> 5;
  const int lane  = tid & 31;
  const int lhalf = lane >> 4;   // 0 or 1 (which 16-lane half)
  const int lcol  = lane & 15;   // N column / A row within tile

  // prologue: sentence lengths, zero h[0] and pool accumulator
  if (tid < ROWS) {
    int cnt = 0;
    const int* dr = docs + (size_t)(base + tid) * S_LEN;
    for (int s = 0; s < S_LEN; ++s) cnt += (dr[s] != 0);
    slen[tid] = cnt;
  }
  {
    uint4 z = make_uint4(0u, 0u, 0u, 0u);
    uint4* p = reinterpret_cast<uint4*>(&hsb[0][0][0]);
    for (int i = tid; i < (ROWS * H_DIM) / 8; i += 256) p[i] = z;
    uint4* q = reinterpret_cast<uint4*>(&ps[0][0]);
    for (int i = tid; i < (ROWS * H_DIM) / 4; i += 256) q[i] = z;
  }

  // preload bias fragments (time-invariant) into registers
  float bias_r[4][4];
  #pragma unroll
  for (int jj = 0; jj < 4; ++jj)
    #pragma unroll
    for (int g = 0; g < 4; ++g)
      bias_r[jj][g] = bias[g * H_DIM + (wave * 4 + jj) * 16 + lcol];

  float creg[4][8];   // cell state fragments (4 hidden tiles per wave)
  #pragma unroll
  for (int j = 0; j < 4; ++j)
    #pragma unroll
    for (int r = 0; r < 8; ++r) creg[j][r] = 0.0f;

  __syncthreads();

  for (int t = 0; t < S_LEN; ++t) {
    const int cur  = t & 1;
    const int nxt  = cur ^ 1;
    const int opos = dir ? (S_LEN - 1 - t) : t;

    // stage x_t tile: 16 rows x 256 bf16; each thread copies 16 elements
    {
      const int row = tid >> 4;
      const int e0  = (tid & 15) * 16;
      const uint4* src = reinterpret_cast<const uint4*>(
          xbf + ((size_t)(base + row) * S_LEN + opos) * E_DIM + e0);
      uint4* dst = reinterpret_cast<uint4*>(&xs[row][e0]);
      dst[0] = src[0];
      dst[1] = src[1];
    }
    __syncthreads();

    #pragma unroll
    for (int jj = 0; jj < 4; ++jj) {
      const int j = wave * 4 + jj;     // hidden-unit tile 0..31

      // Opaque zero offset in an SGPR: defeats LICM so weight loads stay in
      // this pass instead of being hoisted across the 64 time steps and
      // spilled to scratch.
      int wofs = 0;
      asm volatile("" : "+s"(wofs));

      // One per-lane base address per weight matrix for this pass; every
      // fragment load is base + compile-time-constant byte offset, folded
      // into the global_load 24-bit immediate.
      const unsigned voff_ih =
          (unsigned)(((j * 16 + lcol) * E_DIM + lhalf * 16 + wofs) * 2);
      const unsigned voff_hh =
          (unsigned)(((j * 16 + lcol) * H_DIM + lhalf * 16 + wofs) * 2);
      const char* wb_ih = (const char*)wih + voff_ih;
      const char* wb_hh = (const char*)whh + voff_hh;

      // LDS A-fragment base addresses (per-lane), constant k offsets
      const char* ab_x = (const char*)&xs[lcol][0] + lhalf * 32;
      const char* ab_h = (const char*)&hsb[cur][lcol][0] + lhalf * 32;

      // unified k-tile index 0..23: 0..7 -> input (xs/wih), 8..23 -> hidden
      auto loadAk = [&](AF& a, int k) {
        const char* ab = (k < 8) ? (ab_x + k * 64) : (ab_h + (k - 8) * 64);
        a.q[0] = *reinterpret_cast<const uint4*>(ab);
        a.q[1] = *reinterpret_cast<const uint4*>(ab + 16);
      };
      auto loadBk = [&](AF b[4], int k) {
        #pragma unroll
        for (int g = 0; g < 4; ++g) {
          const char* bb = (k < 8)
              ? (wb_ih + g * IH_GSTRIDE + k * 64)
              : (wb_hh + g * HH_GSTRIDE + (k - 8) * 64);
          b[g].q[0] = *reinterpret_cast<const uint4*>(bb);
          b[g].q[1] = *reinterpret_cast<const uint4*>(bb + 16);
        }
      };

      CF acc[4];                       // i, f, g, o accumulators, seeded with bias
      #pragma unroll
      for (int g = 0; g < 4; ++g) {
        const float bv = bias_r[jj][g];
        #pragma unroll
        for (int r = 0; r < 8; ++r) acc[g].f[r] = bv;
      }

      // 2-stage software-pipelined K loop. sched_barrier(0) fences pin the
      // issue order loads(k+1) -> wmma(k) -> loads(k+2) -> wmma(k+1), so the
      // scheduler cannot sink loads back to depth-1 and RA must keep both
      // fragment buffers live: every WMMA group waits with a full group of
      // loads still in flight.
      AF a0, a1;
      AF b0[4], b1[4];
      loadAk(a0, 0);
      loadBk(b0, 0);
      #pragma unroll
      for (int k = 0; k < 24; k += 2) {
        loadAk(a1, k + 1);
        loadBk(b1, k + 1);
        __builtin_amdgcn_sched_barrier(0);
        #pragma unroll
        for (int g = 0; g < 4; ++g)
          acc[g].v = __builtin_amdgcn_wmma_f32_16x16x32_bf16(
              false, a0.v, false, b0[g].v, (short)0, acc[g].v, false, false);
        __builtin_amdgcn_sched_barrier(0);
        if (k + 2 < 24) {
          loadAk(a0, k + 2);
          loadBk(b0, k + 2);
        }
        __builtin_amdgcn_sched_barrier(0);
        #pragma unroll
        for (int g = 0; g < 4; ++g)
          acc[g].v = __builtin_amdgcn_wmma_f32_16x16x32_bf16(
              false, a1.v, false, b1[g].v, (short)0, acc[g].v, false, false);
        __builtin_amdgcn_sched_barrier(0);
      }

      // gate nonlinearities + state update (each (m,n) element single-owner)
      #pragma unroll
      for (int r = 0; r < 8; ++r) {
        const int m  = r + 8 * lhalf;
        const int hcol = j * 16 + lcol;
        const float ig = fsigmoid(acc[0].f[r]);
        const float fg = fsigmoid(acc[1].f[r]);
        const float gg = ftanh_fast(acc[2].f[r]);
        const float og = fsigmoid(acc[3].f[r]);
        const float cn = fg * creg[jj][r] + ig * gg;
        creg[jj][r] = cn;
        const float hn = og * ftanh_fast(cn);
        ps[m][hcol] += (opos < slen[m]) ? hn : 0.0f;
        hsb[nxt][m][hcol] = f2bf(hn);
      }
    }
    __syncthreads();
  }

  // epilogue: mean-pool, doc mask, write output half for this direction
  #pragma unroll
  for (int jj = 0; jj < 4; ++jj) {
    const int j = wave * 4 + jj;
    #pragma unroll
    for (int r = 0; r < 8; ++r) {
      const int m    = r + 8 * lhalf;
      const int nrow = base + m;
      const int bidx = nrow / D_DOCS;
      const int didx = nrow % D_DOCS;
      const int sl   = slen[m];
      const float denom = (float)(sl > 1 ? sl : 1);
      float val = (didx < doc_lens[bidx]) ? (ps[m][j * 16 + lcol] / denom) : 0.0f;
      out[(size_t)nrow * OUT_STRIDE + dir * H_DIM + j * 16 + lcol] = val;
    }
  }
}

// ---------------------------------------------------------------- launch
extern "C" void kernel_launch(void* const* d_in, const int* in_sizes, int n_in,
                              void* d_out, int out_size, void* d_ws, size_t ws_size,
                              hipStream_t stream) {
  (void)in_sizes; (void)n_in; (void)out_size; (void)ws_size;

  const int*   docs     = (const int*)d_in[0];
  const int*   doc_lens = (const int*)d_in[1];
  const float* embed    = (const float*)d_in[3];
  const float* wihf     = (const float*)d_in[4];
  const float* whhf     = (const float*)d_in[5];
  const float* bf       = (const float*)d_in[6];
  const float* wihb     = (const float*)d_in[7];
  const float* whhb     = (const float*)d_in[8];
  const float* bb       = (const float*)d_in[9];
  float*       out      = (float*)d_out;

  const int NTOK  = 32 * 32 * S_LEN;          // 65536 tokens
  const int N_WIH = G_DIM * E_DIM;            // 524288
  const int N_WHH = G_DIM * H_DIM;            // 1048576

  char* ws = (char*)d_ws;
  unsigned short* xbf    = (unsigned short*)ws;  ws += (size_t)NTOK * E_DIM * 2;
  unsigned short* wihf16 = (unsigned short*)ws;  ws += (size_t)N_WIH * 2;
  unsigned short* whhf16 = (unsigned short*)ws;  ws += (size_t)N_WHH * 2;
  unsigned short* wihb16 = (unsigned short*)ws;  ws += (size_t)N_WIH * 2;
  unsigned short* whhb16 = (unsigned short*)ws;

  cvt_bf16_kernel<<<(N_WIH + 255) / 256, 256, 0, stream>>>(wihf, wihf16, N_WIH);
  cvt_bf16_kernel<<<(N_WHH + 255) / 256, 256, 0, stream>>>(whhf, whhf16, N_WHH);
  cvt_bf16_kernel<<<(N_WIH + 255) / 256, 256, 0, stream>>>(wihb, wihb16, N_WIH);
  cvt_bf16_kernel<<<(N_WHH + 255) / 256, 256, 0, stream>>>(whhb, whhb16, N_WHH);

  embed_kernel<<<NTOK / 8, 256, 0, stream>>>(docs, embed, xbf);

  dim3 grid(64, 2);
  lstm_kernel<<<grid, 256, 0, stream>>>(xbf, docs, doc_lens,
                                        wihf16, whhf16, wihb16, whhb16,
                                        bf, bb, out);
}
